// DistGAT_7533372637271
// MI455X (gfx1250) — matrix operand order repaired and checked by reading, not compiled, not run on hardware
//
#include <hip/hip_runtime.h>
#include <hip/hip_bf16.h>
#include <math.h>

// ---------------- problem constants (match reference) ----------------
#define NNODES 50000
#define NEDGES 800000
#define IN_DIM 128
#define HID    32
#define OUTD   64
#define H1     4
#define H2     4
#define H3     6
#define NEG_SLOPE 0.2f

typedef __attribute__((ext_vector_type(16))) _Float16 v16h;
typedef __attribute__((ext_vector_type(8)))  _Float16 v8h;
typedef __attribute__((ext_vector_type(8)))  float    v8f;

// ---------------- helpers ----------------
__device__ __forceinline__ float atomicMaxF(float* addr, float val) {
    // classic ordered-float trick; buffer initialized to -inf
    if (val >= 0.0f)
        return __int_as_float(atomicMax((int*)addr, __float_as_int(val)));
    else
        return __uint_as_float(atomicMin((unsigned int*)addr, __float_as_uint(val)));
}

// ---------------- elementwise / setup kernels ----------------
__global__ void fill_f32_k(float* __restrict__ p, float v, int n) {
    int i = blockIdx.x * blockDim.x + threadIdx.x;
    if (i < n) p[i] = v;
}

__global__ void cvt_f32_f16_k(const float* __restrict__ in, _Float16* __restrict__ out, int n) {
    int i = blockIdx.x * blockDim.x + threadIdx.x;
    if (i < n) out[i] = (_Float16)in[i];
}

// W[k*C + c] (f32) -> Wt[c*K + k] (f16), so each WMMA B-column is contiguous in K
__global__ void cvt_transpose_k(const float* __restrict__ W, _Float16* __restrict__ Wt,
                                int K, int C) {
    int i = blockIdx.x * blockDim.x + threadIdx.x;
    if (i < K * C) {
        int k = i / C, c = i - k * C;
        Wt[(size_t)c * K + k] = (_Float16)W[i];
    }
}

// ---------------- WMMA GEMM: C[M,Ncols] = A[M,K] * B[K,Ncols] ----------------
// A: f16 row-major [M,K]; Bt: f16 [Ncols,K] (B transposed); M % 16 == 0, K == 128,
// Ncols % 64 == 0. One wave computes a 16x64 strip (4 x 16x16 tiles) so the A
// fragment is loaded once per k-step and reused across 4 WMMAs.
__global__ void gemm_wmma_f16_k(const _Float16* __restrict__ A,
                                const _Float16* __restrict__ Bt,
                                float* __restrict__ C,
                                int M, int K, int Ncols) {
    int wave = (int)((blockIdx.x * (long long)blockDim.x + threadIdx.x) >> 5);
    int lane = threadIdx.x & 31;
    int stripsN = Ncols >> 6;            // 16x64 strips along N
    int tM = wave / stripsN;
    int sN = wave - tM * stripsN;
    if (tM * 16 >= M) return;

    int hi = lane >> 4;     // 0: lanes 0-15, 1: lanes 16-31
    int l  = lane & 15;

    const _Float16* arow = A + (size_t)(tM * 16 + l) * K;           // A row for this lane
    const _Float16* brow0 = Bt + (size_t)(sN * 64 + l) * K;          // B columns, 4 tiles
    size_t bTileStride = (size_t)16 * K;                             // 16 columns apart

    v8f acc0 = {}, acc1 = {}, acc2 = {}, acc3 = {};
    #pragma unroll
    for (int k0 = 0; k0 < 128; k0 += 32) {
        // A fragment (16x32 f16): lanes 0-15 hold K = k0..k0+15,
        //                          lanes 16-31 hold K = k0+16..k0+31
        union { v16h v; v8h h2[2]; } a;
        a.h2[0] = *(const v8h*)(arow + k0 + hi * 16);
        a.h2[1] = *(const v8h*)(arow + k0 + 8 + hi * 16);
        // B fragments (32x16 f16): lanes 0-15 hold K = k0..k0+15, lanes 16-31 K = k0+16..k0+31
        v16h b0 = *(const v16h*)(brow0 + 0 * bTileStride + k0 + hi * 16);
        v16h b1 = *(const v16h*)(brow0 + 1 * bTileStride + k0 + hi * 16);
        v16h b2 = *(const v16h*)(brow0 + 2 * bTileStride + k0 + hi * 16);
        v16h b3 = *(const v16h*)(brow0 + 3 * bTileStride + k0 + hi * 16);
        acc0 = __builtin_amdgcn_wmma_f32_16x16x32_f16(false, a.v, false, b0, (short)0, acc0, false, false);
        acc1 = __builtin_amdgcn_wmma_f32_16x16x32_f16(false, a.v, false, b1, (short)0, acc1, false, false);
        acc2 = __builtin_amdgcn_wmma_f32_16x16x32_f16(false, a.v, false, b2, (short)0, acc2, false, false);
        acc3 = __builtin_amdgcn_wmma_f32_16x16x32_f16(false, a.v, false, b3, (short)0, acc3, false, false);
    }
    // C/D layout: VGPR r -> row tM*16 + hi*8 + r, col base + l
    union { v8f v; float f[8]; } r0, r1, r2, r3;
    r0.v = acc0; r1.v = acc1; r2.v = acc2; r3.v = acc3;
    float* cbase = C + (size_t)(tM * 16 + hi * 8) * Ncols + sN * 64 + l;
    #pragma unroll
    for (int i = 0; i < 8; ++i) {
        float* row = cbase + (size_t)i * Ncols;
        row[0]  = r0.f[i];
        row[16] = r1.f[i];
        row[32] = r2.f[i];
        row[48] = r3.f[i];
    }
}

// ---------------- attention coefficients: el/er = sum_d feat*al / feat*ar ----------------
__global__ void attn_coeff_k(const float* __restrict__ feat, const float* __restrict__ al,
                             const float* __restrict__ ar, float* __restrict__ el,
                             float* __restrict__ er, int N, int H, int D) {
    int wave = (int)((blockIdx.x * (long long)blockDim.x + threadIdx.x) >> 5);
    int lane = threadIdx.x & 31;
    if (wave >= N) return;
    const float* f = feat + (size_t)wave * H * D;
    for (int h = 0; h < H; ++h) {
        float pl = 0.0f, pr = 0.0f;
        for (int d = lane; d < D; d += 32) {
            float v = f[h * D + d];
            pl += v * al[h * D + d];
            pr += v * ar[h * D + d];
        }
        #pragma unroll
        for (int off = 16; off > 0; off >>= 1) {
            pl += __shfl_xor(pl, off, 32);
            pr += __shfl_xor(pr, off, 32);
        }
        if (lane == 0) { el[wave * H + h] = pl; er[wave * H + h] = pr; }
    }
}

// ---------------- edge passes (segment softmax over dst, all through L2 atomics) -------
__global__ void edge_max_k(const int* __restrict__ src, const int* __restrict__ dst,
                           const float* __restrict__ el, const float* __restrict__ er,
                           float* __restrict__ m, int E, int H) {
    int t = blockIdx.x * blockDim.x + threadIdx.x;
    if (t >= E * H) return;
    int e = t / H, h = t - e * H;
    int s = src[e], d = dst[e];
    float x = el[s * H + h] + er[d * H + h];
    x = (x > 0.0f) ? x : x * NEG_SLOPE;
    atomicMaxF(&m[d * H + h], x);
}

__global__ void edge_expsum_k(const int* __restrict__ src, const int* __restrict__ dst,
                              const float* __restrict__ el, const float* __restrict__ er,
                              const float* __restrict__ m, float* __restrict__ ex,
                              float* __restrict__ ssum, int E, int H) {
    int t = blockIdx.x * blockDim.x + threadIdx.x;
    if (t >= E * H) return;
    int e = t / H, h = t - e * H;
    int s = src[e], d = dst[e];
    float x = el[s * H + h] + er[d * H + h];
    x = (x > 0.0f) ? x : x * NEG_SLOPE;
    float v = expf(x - m[d * H + h]);
    ex[t] = v;
    atomicAdd(&ssum[d * H + h], v);
}

// one thread per (edge, head, d); coalesced gather of feat[src], coalesced L2 atomic scatter
__global__ void edge_aggregate_k(const int* __restrict__ src, const int* __restrict__ dst,
                                 const float* __restrict__ feat, const float* __restrict__ ex,
                                 const float* __restrict__ ssum, float* __restrict__ rst,
                                 int E, int H, int D) {
    int t = blockIdx.x * blockDim.x + threadIdx.x;
    if (t >= E * H * D) return;
    int d  = t % D;
    int eh = t / D;
    int h  = eh % H;
    int e  = eh / H;
    int sN = src[e], dN = dst[e];
    float a = ex[e * H + h] / ssum[dN * H + h];
    float v = feat[(size_t)sN * H * D + h * D + d] * a;
    atomicAdd(&rst[(size_t)dN * H * D + h * D + d], v);
}

// ---------------- finalize ----------------
__global__ void finalize_elu_k(const float* __restrict__ rst, const float* __restrict__ b,
                               float* __restrict__ out, int N, int HD) {
    int t = blockIdx.x * blockDim.x + threadIdx.x;
    if (t >= N * HD) return;
    int i = t % HD;
    float x = rst[t] + b[i];
    out[t] = (x > 0.0f) ? x : (expf(x) - 1.0f);          // elu, alpha=1
}

__global__ void finalize_mean_k(const float* __restrict__ rst, const float* __restrict__ b,
                                float* __restrict__ out, int N, int H, int D) {
    int t = blockIdx.x * blockDim.x + threadIdx.x;
    if (t >= N * D) return;
    int n = t / D, o = t - n * D;
    float acc = 0.0f;
    for (int h = 0; h < H; ++h)
        acc += rst[(size_t)n * H * D + h * D + o] + b[h * D + o];
    out[t] = acc / (float)H;
}

// ---------------- host launcher ----------------
static inline int cdiv(long long a, int b) { return (int)((a + b - 1) / b); }

extern "C" void kernel_launch(void* const* d_in, const int* in_sizes, int n_in,
                              void* d_out, int out_size, void* d_ws, size_t ws_size,
                              hipStream_t stream) {
    (void)in_sizes; (void)n_in; (void)out_size; (void)ws_size;
    const float* x    = (const float*)d_in[0];
    const int*   src  = (const int*)d_in[1];
    const int*   dst  = (const int*)d_in[2];
    const float* W1   = (const float*)d_in[3];
    const float* al1  = (const float*)d_in[4];
    const float* ar1  = (const float*)d_in[5];
    const float* b1   = (const float*)d_in[6];
    const float* W2   = (const float*)d_in[7];
    const float* al2  = (const float*)d_in[8];
    const float* ar2  = (const float*)d_in[9];
    const float* b2   = (const float*)d_in[10];
    const float* W3   = (const float*)d_in[11];
    const float* al3  = (const float*)d_in[12];
    const float* ar3  = (const float*)d_in[13];
    const float* b3   = (const float*)d_in[14];
    const float* Wres3= (const float*)d_in[15];
    float* out = (float*)d_out;

    // ---- carve workspace (256B aligned slices) ----
    char* ws = (char*)d_ws;
    auto carve = [&](size_t bytes) -> char* {
        char* p = ws;
        ws += (bytes + 255) & ~(size_t)255;
        return p;
    };
    _Float16* Af16 = (_Float16*)carve((size_t)NNODES * IN_DIM * sizeof(_Float16)); // 12.8MB
    _Float16* Wt   = (_Float16*)carve((size_t)384 * 128 * sizeof(_Float16));       // 96KB
    float* feat = (float*)carve((size_t)NNODES * 384 * sizeof(float));             // 76.8MB
    float* rst  = (float*)carve((size_t)NNODES * 384 * sizeof(float));             // 76.8MB
    float* hbuf = (float*)carve((size_t)NNODES * IN_DIM * sizeof(float));          // 25.6MB
    float* el   = (float*)carve((size_t)NNODES * H3 * sizeof(float));
    float* er   = (float*)carve((size_t)NNODES * H3 * sizeof(float));
    float* mbuf = (float*)carve((size_t)NNODES * H3 * sizeof(float));
    float* sbuf = (float*)carve((size_t)NNODES * H3 * sizeof(float));
    float* ex   = (float*)carve((size_t)NEDGES * H3 * sizeof(float));              // 19.2MB

    const int B = 256;
    const int rowTiles = NNODES / 16;  // 3125 exactly

    // ================= layer 1 : IN -> H1*HID, residual none, elu =================
    {
        const int H = H1, D = HID, NC = H * D; // 128
        cvt_f32_f16_k<<<cdiv((long long)NNODES * IN_DIM, B), B, 0, stream>>>(x, Af16, NNODES * IN_DIM);
        cvt_transpose_k<<<cdiv((long long)IN_DIM * NC, B), B, 0, stream>>>(W1, Wt, IN_DIM, NC);
        {
            long long waves = (long long)rowTiles * (NC / 64);
            gemm_wmma_f16_k<<<cdiv(waves * 32, B), B, 0, stream>>>(Af16, Wt, feat, NNODES, IN_DIM, NC);
        }
        attn_coeff_k<<<cdiv((long long)NNODES * 32, B), B, 0, stream>>>(feat, al1, ar1, el, er, NNODES, H, D);
        fill_f32_k<<<cdiv((long long)NNODES * H, B), B, 0, stream>>>(mbuf, -INFINITY, NNODES * H);
        fill_f32_k<<<cdiv((long long)NNODES * H, B), B, 0, stream>>>(sbuf, 0.0f, NNODES * H);
        fill_f32_k<<<cdiv((long long)NNODES * NC, B), B, 0, stream>>>(rst, 0.0f, NNODES * NC);
        edge_max_k<<<cdiv((long long)NEDGES * H, B), B, 0, stream>>>(src, dst, el, er, mbuf, NEDGES, H);
        edge_expsum_k<<<cdiv((long long)NEDGES * H, B), B, 0, stream>>>(src, dst, el, er, mbuf, ex, sbuf, NEDGES, H);
        edge_aggregate_k<<<cdiv((long long)NEDGES * H * D, B), B, 0, stream>>>(src, dst, feat, ex, sbuf, rst, NEDGES, H, D);
        finalize_elu_k<<<cdiv((long long)NNODES * NC, B), B, 0, stream>>>(rst, b1, hbuf, NNODES, NC);
    }

    // ================= layer 2 : identity residual, elu =================
    {
        const int H = H2, D = HID, NC = H * D; // 128
        cvt_f32_f16_k<<<cdiv((long long)NNODES * NC, B), B, 0, stream>>>(hbuf, Af16, NNODES * NC);
        cvt_transpose_k<<<cdiv((long long)NC * NC, B), B, 0, stream>>>(W2, Wt, NC, NC);
        {
            long long waves = (long long)rowTiles * (NC / 64);
            gemm_wmma_f16_k<<<cdiv(waves * 32, B), B, 0, stream>>>(Af16, Wt, feat, NNODES, NC, NC);
        }
        attn_coeff_k<<<cdiv((long long)NNODES * 32, B), B, 0, stream>>>(feat, al2, ar2, el, er, NNODES, H, D);
        fill_f32_k<<<cdiv((long long)NNODES * H, B), B, 0, stream>>>(mbuf, -INFINITY, NNODES * H);
        fill_f32_k<<<cdiv((long long)NNODES * H, B), B, 0, stream>>>(sbuf, 0.0f, NNODES * H);
        // residual folded into rst initialization: rst = h
        hipMemcpyAsync(rst, hbuf, (size_t)NNODES * NC * sizeof(float),
                       hipMemcpyDeviceToDevice, stream);
        edge_max_k<<<cdiv((long long)NEDGES * H, B), B, 0, stream>>>(src, dst, el, er, mbuf, NEDGES, H);
        edge_expsum_k<<<cdiv((long long)NEDGES * H, B), B, 0, stream>>>(src, dst, el, er, mbuf, ex, sbuf, NEDGES, H);
        edge_aggregate_k<<<cdiv((long long)NEDGES * H * D, B), B, 0, stream>>>(src, dst, feat, ex, sbuf, rst, NEDGES, H, D);
        finalize_elu_k<<<cdiv((long long)NNODES * NC, B), B, 0, stream>>>(rst, b2, hbuf, NNODES, NC);
    }

    // ================= layer 3 : projected residual, mean over heads =================
    {
        const int H = H3, D = OUTD, NC = H * D; // 384
        const int KIN = H2 * HID;               // 128
        cvt_f32_f16_k<<<cdiv((long long)NNODES * KIN, B), B, 0, stream>>>(hbuf, Af16, NNODES * KIN);
        // feat = h2 @ W3
        cvt_transpose_k<<<cdiv((long long)KIN * NC, B), B, 0, stream>>>(W3, Wt, KIN, NC);
        {
            long long waves = (long long)rowTiles * (NC / 64);
            gemm_wmma_f16_k<<<cdiv(waves * 32, B), B, 0, stream>>>(Af16, Wt, feat, NNODES, KIN, NC);
        }
        // residual folded into rst initialization: rst = h2 @ Wres3
        cvt_transpose_k<<<cdiv((long long)KIN * NC, B), B, 0, stream>>>(Wres3, Wt, KIN, NC);
        {
            long long waves = (long long)rowTiles * (NC / 64);
            gemm_wmma_f16_k<<<cdiv(waves * 32, B), B, 0, stream>>>(Af16, Wt, rst, NNODES, KIN, NC);
        }
        attn_coeff_k<<<cdiv((long long)NNODES * 32, B), B, 0, stream>>>(feat, al3, ar3, el, er, NNODES, H, D);
        fill_f32_k<<<cdiv((long long)NNODES * H, B), B, 0, stream>>>(mbuf, -INFINITY, NNODES * H);
        fill_f32_k<<<cdiv((long long)NNODES * H, B), B, 0, stream>>>(sbuf, 0.0f, NNODES * H);
        edge_max_k<<<cdiv((long long)NEDGES * H, B), B, 0, stream>>>(src, dst, el, er, mbuf, NEDGES, H);
        edge_expsum_k<<<cdiv((long long)NEDGES * H, B), B, 0, stream>>>(src, dst, el, er, mbuf, ex, sbuf, NEDGES, H);
        edge_aggregate_k<<<cdiv((long long)NEDGES * H * D, B), B, 0, stream>>>(src, dst, feat, ex, sbuf, rst, NEDGES, H, D);
        finalize_mean_k<<<cdiv((long long)NNODES * D, B), B, 0, stream>>>(rst, b3, out, NNODES, H, D);
    }
}
